// CausalSelfAttentionOrig_41446434406867
// MI455X (gfx1250) — compile-verified, matrix operand order checked by reading, not run on hardware
//
#include <hip/hip_runtime.h>
#include <hip/hip_bf16.h>

#ifndef __has_builtin
#define __has_builtin(x) 0
#endif

#if __has_builtin(__builtin_amdgcn_global_load_async_to_lds_b128)
#define USE_ASYNC_LDS 1
#else
#define USE_ASYNC_LDS 0
#endif

#define AS_GLOBAL __attribute__((address_space(1)))
#define AS_LOCAL  __attribute__((address_space(3)))

typedef __attribute__((ext_vector_type(16))) __bf16 v16bf;
typedef __attribute__((ext_vector_type(8)))  float  v8f;
typedef __attribute__((ext_vector_type(4)))  int    v4i;

// ---------------------------------------------------------------------------
// WMMA helper: D = A(16x32 bf16) * B(32x16 bf16) + C(16x16 f32)
// ---------------------------------------------------------------------------
__device__ __forceinline__ v8f wmma_bf16(v16bf a, v16bf b, v8f c) {
    return __builtin_amdgcn_wmma_f32_16x16x32_bf16(
        /*neg_a=*/false, a, /*neg_b=*/false, b,
        /*c_mod=*/(short)0, c, /*reuse_a=*/false, /*reuse_b=*/false);
}

// Per-lane A/B fragment K-map (16-bit 16x32 layout, ISA 7.12.2):
//   lanes 0-15 : K = {0..7, 16..23};  lanes 16-31 : K = {8..15, 24..31}
// => two contiguous 8-element runs starting at (hi*8) and (16 + hi*8).

// bf16 source (global or LDS), 16-byte aligned rows -> two 128b loads
__device__ __forceinline__ v16bf ldrow_bf16(const __bf16* p, int hi) {
    union { v16bf v; uint4 u[2]; } r;
    r.u[0] = *(const uint4*)(p + hi * 8);
    r.u[1] = *(const uint4*)(p + 16 + hi * 8);
    return r.v;
}

// f32 source row (contiguous along K), convert to bf16
__device__ __forceinline__ v16bf ldrow_f32(const float* p, int hi) {
    v16bf f;
#pragma unroll
    for (int i = 0; i < 8; ++i) {
        f[i]     = (__bf16)p[hi * 8 + i];
        f[8 + i] = (__bf16)p[16 + hi * 8 + i];
    }
    return f;
}

// ---------------------------------------------------------------------------
// Generic GEMM: out[M,N] = A[M,K] * W[K,N] + bias[N]
// Block = 256 threads (8 waves). Wave w -> rows [blockIdx.y*128 + w*16, +16),
// block cols [blockIdx.x*64, +64). Per 64-wide k-step: W tile (64K x 64N) is
// cooperatively staged transposed (N x K) in LDS as bf16; each wave then does
// 2 A-frags + 8 LDS B-frags + 8 WMMAs.
// ---------------------------------------------------------------------------
template <bool A_BF16, bool OUT_BF16>
__global__ __launch_bounds__(256) void gemm_wmma_kernel(
    const void* __restrict__ Ap, const float* __restrict__ W,
    const float* __restrict__ bias, void* __restrict__ outp,
    int M, int N, int K) {
    const int lane = threadIdx.x & 31;
    const int wave = threadIdx.x >> 5;
    const int hi   = lane >> 4;
    const int l16  = lane & 15;

    const int rowBase = blockIdx.y * 128 + wave * 16;
    const int nBase   = blockIdx.x * 64;

    const float*  Af = (const float*)Ap;
    const __bf16* Ab = (const __bf16*)Ap;
    const int arow = rowBase + l16;

    __shared__ __align__(16) __bf16 Wt[64][72];   // N x K(64), padded

    // staging role: thread -> (k row, 16-wide n chunk)
    const int sk = threadIdx.x >> 2;        // 0..63
    const int sn = (threadIdx.x & 3) * 16;  // 0,16,32,48

    v8f acc[4] = {};

    for (int kb = 0; kb < K; kb += 64) {
        // ---- stage W tile (f32 -> bf16, transposed) ----
        {
            const float* wsrc = W + (size_t)(kb + sk) * N + nBase + sn;
            if (kb + 64 < K)
                __builtin_prefetch(wsrc + (size_t)64 * N, 0, 3);
#pragma unroll
            for (int i = 0; i < 16; ++i)
                Wt[sn + i][sk] = (__bf16)wsrc[i];
        }
        __syncthreads();

        // ---- A fragments (dims kb..kb+31, kb+32..kb+63) ----
        v16bf a0, a1;
        if (A_BF16) {
            a0 = ldrow_bf16(Ab + (size_t)arow * K + kb,      hi);
            a1 = ldrow_bf16(Ab + (size_t)arow * K + kb + 32, hi);
        } else {
            a0 = ldrow_f32(Af + (size_t)arow * K + kb,      hi);
            a1 = ldrow_f32(Af + (size_t)arow * K + kb + 32, hi);
        }

#pragma unroll
        for (int t = 0; t < 4; ++t) {
            v16bf b0 = ldrow_bf16(&Wt[t * 16 + l16][0],  hi);
            v16bf b1 = ldrow_bf16(&Wt[t * 16 + l16][32], hi);
            acc[t] = wmma_bf16(a0, b0, acc[t]);
            acc[t] = wmma_bf16(a1, b1, acc[t]);
        }
        __syncthreads();
    }

#pragma unroll
    for (int t = 0; t < 4; ++t) {
        const int n  = nBase + t * 16 + l16;
        const float bv = bias[n];
#pragma unroll
        for (int r = 0; r < 8; ++r) {
            const int row = rowBase + r + hi * 8;
            const float v = acc[t][r] + bv;
            if (OUT_BF16) ((__bf16*)outp)[(size_t)row * N + n] = (__bf16)v;
            else          ((float*)outp)[(size_t)row * N + n]  = v;
        }
    }
}

// ---------------------------------------------------------------------------
// Flash attention (causal), bf16 QKV in, bf16 out.
// Grid: (T/128, H, B). Block 256 = 8 waves; wave w owns q rows [qblock+w*16, +16).
// Key tiles of 32 staged in LDS: K via async global->LDS copies (ASYNCcnt),
// V transposed by threads. Online softmax with 16-lane shuffle reductions.
// ---------------------------------------------------------------------------
__global__ __launch_bounds__(256) void flash_attn_kernel(
    const __bf16* __restrict__ qkv, __bf16* __restrict__ y,
    int T, int C, int H) {
    const int D    = 64;
    const int lane = threadIdx.x & 31;
    const int wave = threadIdx.x >> 5;
    const int hi   = lane >> 4;
    const int l16  = lane & 15;

    const int b      = blockIdx.z;
    const int h      = blockIdx.y;
    const int qblock = blockIdx.x * 128;
    const int qb     = qblock + wave * 16;

    const size_t rstride = (size_t)3 * C;   // qkv row stride (elements)

    __shared__ __align__(16) __bf16 Kl[32][72];      // keys x dims (padded)
    __shared__ __align__(16) __bf16 Vt[64][40];      // dims x keys (transposed, padded)
    __shared__ __align__(16) __bf16 Pl[8][16][40];   // per-wave P scratch

    // persistent Q fragments (dims 0..31 and 32..63), row = qb + l16
    const __bf16* qrow = qkv + ((size_t)b * T + qb + l16) * rstride + h * D;
    const v16bf q0 = ldrow_bf16(qrow,      hi);
    const v16bf q1 = ldrow_bf16(qrow + 32, hi);

    v8f  o[4] = {};
    float m[8], l[8];
#pragma unroll
    for (int r = 0; r < 8; ++r) { m[r] = -1e30f; l[r] = 0.0f; }

    const float scale  = 0.03125f;          // 1/sqrt(C) = 1/32
    const int   ktiles = qblock / 32 + 4;   // causal upper bound for this block

    for (int kt = 0; kt < ktiles; ++kt) {
        // ---- stage K (async global->LDS) and V (transposed) tiles ----
        {
            const int key = threadIdx.x >> 3;          // 0..31
            const int dc  = (threadIdx.x & 7) * 8;     // 0..56 step 8
            const __bf16* ks = qkv + ((size_t)b * T + kt * 32 + key) * rstride + C + h * D + dc;
            const __bf16* vs = ks + C;
#if USE_ASYNC_LDS
            __builtin_amdgcn_global_load_async_to_lds_b128(
                (AS_GLOBAL v4i*)ks, (AS_LOCAL v4i*)&Kl[key][dc],
                /*offset=*/0, /*cpol=*/0);
#else
            *(uint4*)(&Kl[key][dc]) = *(const uint4*)ks;
#endif
            union { uint4 u; __bf16 e[8]; } vv;
            vv.u = *(const uint4*)vs;
#pragma unroll
            for (int i = 0; i < 8; ++i) Vt[dc + i][key] = vv.e[i];
#if USE_ASYNC_LDS
            asm volatile("s_wait_asynccnt 0" ::: "memory");
#endif
        }
        __syncthreads();

        // ---- S = Q * K^T  (two 16-key halves) ----
        v8f s0 = {}, s1 = {};
        {
            v16bf b0 = ldrow_bf16(&Kl[l16][0],  hi);
            v16bf b1 = ldrow_bf16(&Kl[l16][32], hi);
            s0 = wmma_bf16(q0, b0, s0);
            s0 = wmma_bf16(q1, b1, s0);
            v16bf c0 = ldrow_bf16(&Kl[16 + l16][0],  hi);
            v16bf c1 = ldrow_bf16(&Kl[16 + l16][32], hi);
            s1 = wmma_bf16(q0, c0, s1);
            s1 = wmma_bf16(q1, c1, s1);
        }

        // ---- online softmax + write P to LDS in C-layout ----
#pragma unroll
        for (int r = 0; r < 8; ++r) {
            const int rowq = qb + r + hi * 8;
            const int key0 = kt * 32 + l16;
            const int key1 = key0 + 16;
            float a0 = (key0 <= rowq) ? s0[r] * scale : -1e30f;
            float a1 = (key1 <= rowq) ? s1[r] * scale : -1e30f;
            float vmax = fmaxf(a0, a1);
#pragma unroll
            for (int off = 8; off; off >>= 1)
                vmax = fmaxf(vmax, __shfl_xor(vmax, off, 16));
            const float mn    = fmaxf(m[r], vmax);
            const float alpha = __expf(m[r] - mn);
            const float p0 = __expf(a0 - mn);
            const float p1 = __expf(a1 - mn);
            float rs = p0 + p1;
#pragma unroll
            for (int off = 8; off; off >>= 1)
                rs += __shfl_xor(rs, off, 16);
            l[r] = l[r] * alpha + rs;
            m[r] = mn;
#pragma unroll
            for (int t = 0; t < 4; ++t) o[t][r] *= alpha;
            const int prow = r + hi * 8;
            Pl[wave][prow][l16]      = (__bf16)p0;
            Pl[wave][prow][16 + l16] = (__bf16)p1;
        }
        asm volatile("s_wait_dscnt 0" ::: "memory");   // wave-local LDS RAW

        // ---- O += P * V  (A-frag from LDS scratch, B-frags from transposed V) ----
        const v16bf pf = ldrow_bf16(&Pl[wave][l16][0], hi);
#pragma unroll
        for (int t = 0; t < 4; ++t) {
            v16bf vb = ldrow_bf16(&Vt[t * 16 + l16][0], hi);
            o[t] = wmma_bf16(pf, vb, o[t]);
        }
        __syncthreads();
    }

    // ---- normalize and store (bf16, [B,T,C] with head offset) ----
#pragma unroll
    for (int r = 0; r < 8; ++r) {
        const float inv  = 1.0f / l[r];
        const int   rowq = qb + r + hi * 8;
        __bf16* dst = y + ((size_t)b * T + rowq) * (size_t)C + h * D;
#pragma unroll
        for (int t = 0; t < 4; ++t)
            dst[t * 16 + l16] = (__bf16)(o[t][r] * inv);
    }
}

// ---------------------------------------------------------------------------
// Host-side launch
// ---------------------------------------------------------------------------
extern "C" void kernel_launch(void* const* d_in, const int* in_sizes, int n_in,
                              void* d_out, int out_size, void* d_ws, size_t ws_size,
                              hipStream_t stream) {
    constexpr int B = 2, T = 2048, C = 1024, H = 16;
    const float* x      = (const float*)d_in[0];
    const float* W_attn = (const float*)d_in[1];
    const float* b_attn = (const float*)d_in[2];
    const float* W_proj = (const float*)d_in[3];
    const float* b_proj = (const float*)d_in[4];
    float* out = (float*)d_out;

    __bf16* qkv = (__bf16*)d_ws;                       // [B*T, 3C] bf16
    __bf16* y   = qkv + (size_t)B * T * 3 * C;         // [B*T, C]  bf16

    // 1) qkv = x @ W_attn + b_attn   (M=4096, N=3072, K=1024), bf16 out
    {
        dim3 grid(3 * C / 64, B * T / 128);
        gemm_wmma_kernel<false, true><<<grid, 256, 0, stream>>>(
            x, W_attn, b_attn, qkv, B * T, 3 * C, C);
    }
    // 2) flash attention -> y (bf16)
    {
        dim3 grid(T / 128, H, B);
        flash_attn_kernel<<<grid, 256, 0, stream>>>(qkv, y, T, C, H);
    }
    // 3) out = y @ W_proj + b_proj   (M=4096, N=1024, K=1024), f32 out
    {
        dim3 grid(C / 64, B * T / 128);
        gemm_wmma_kernel<true, false><<<grid, 256, 0, stream>>>(
            y, W_proj, b_proj, out, B * T, C, C);
    }
}
